// SparseAutoencoder_68092411511062
// MI455X (gfx1250) — matrix operand order repaired and checked by reading, not compiled, not run on hardware
//
#include <hip/hip_runtime.h>
#include <hip/hip_bf16.h>

#define INPUT_DIM  4096
#define HIDDEN_DIM 16384
#define BATCH      8192
#define TOPK       32

#define XS_STRIDE   4104           // 4096 halfs + 16B pad -> conflict-free b128 LDS reads
#define TILE_STRIDE 68
#define BROW_BYTES  528u           // 256 bf16 (512B) + 16B pad per B row
#define BWAVE_BYTES (16u * BROW_BYTES)      // one wave's 16 B rows  = 8448 B
#define BBUF_BYTES  (4u * BWAVE_BYTES)      // all 4 waves, one buffer = 33792 B
#define BST0        ((unsigned)(16 * XS_STRIDE * 2))   // B-stage base = 131328

typedef __attribute__((ext_vector_type(16))) __bf16         v16bf;
typedef __attribute__((ext_vector_type(8)))  float          v8f;
typedef __attribute__((ext_vector_type(8)))  unsigned short u16x8;
typedef __attribute__((ext_vector_type(4)))  unsigned short u16x4;

__device__ __forceinline__ unsigned short f32_to_bf16(float f) {
    union { float f; unsigned int u; } c; c.f = f;
    unsigned int u = c.u;
    u += 0x7FFFu + ((u >> 16) & 1u);      // round-to-nearest-even
    return (unsigned short)(u >> 16);
}
__device__ __forceinline__ float bf16_to_f32(unsigned short h) {
    union { unsigned int u; float f; } c; c.u = ((unsigned int)h) << 16;
    return c.f;
}

// CDNA5 async global->LDS copy (ASYNCcnt-tracked). vdst VGPR = LDS byte addr,
// GVS addressing: mem = SADDR(64) + VADDR(32).
__device__ __forceinline__ void async_load_b128(unsigned lds_byte_addr,
                                                const void* sbase,
                                                unsigned voff_bytes) {
    asm volatile("global_load_async_to_lds_b128 %0, %1, %2"
                 :
                 : "v"(lds_byte_addr), "v"(voff_bytes), "s"(sbase)
                 : "memory");
}
template <int N>
__device__ __forceinline__ void wait_asynccnt() {
    asm volatile("s_wait_asynccnt %0" :: "i"(N) : "memory");
}

// ---------------------------------------------------------------------------
// Kernel 0: W_enc f32 -> bf16 (128 MB copy: fits MI455X 192 MB L2 afterwards)
// ---------------------------------------------------------------------------
__global__ void wenc_to_bf16(const float* __restrict__ W_enc,
                             unsigned short* __restrict__ WencB) {
    size_t e4 = (size_t)blockIdx.x * blockDim.x + threadIdx.x;   // float4 index
    float4 v = ((const float4*)W_enc)[e4];
    u16x4 p;
    p.x = f32_to_bf16(v.x); p.y = f32_to_bf16(v.y);
    p.z = f32_to_bf16(v.z); p.w = f32_to_bf16(v.w);
    *(u16x4*)&WencB[e4 * 4] = p;
}

// ---------------------------------------------------------------------------
// Kernel 1: fused encode GEMM (bf16 WMMA, async-LDS double-buffered B) +
// streaming per-row top-K. Block = 128 threads (4 waves) = 16 batch rows.
// ---------------------------------------------------------------------------
__global__ void sae_encode_topk(const float* __restrict__ x,
                                const unsigned short* __restrict__ WencB,
                                const float* __restrict__ b_enc,
                                float* __restrict__ top_vals,
                                int* __restrict__ top_idx) {
    extern __shared__ unsigned char smem_raw[];
    unsigned short* xs     = (unsigned short*)smem_raw;            // [16][XS_STRIDE]
    float* tile   = (float*)(smem_raw + BST0 + 2 * BBUF_BYTES);    // [16][TILE_STRIDE]
    float* topv   = tile + 16 * TILE_STRIDE;                       // [16][TOPK]
    int*   topi   = (int*)(topv + 16 * TOPK);                      // [16][TOPK]
    float* rowmin = (float*)(topi + 16 * TOPK);                    // [16]
    int*   rowpos = (int*)(rowmin + 16);                           // [16]

    const int tid  = threadIdx.x;
    const int wave = tid >> 5;
    const int lane = tid & 31;
    const int row0 = blockIdx.x * 16;

    // -------- Phase 0: stage x tile (f32 -> bf16) into LDS, init top-k state
    const float4* xg = (const float4*)(x + (size_t)row0 * INPUT_DIM);
    #pragma unroll 4
    for (int it = 0; it < (16 * INPUT_DIM / 4) / 128; ++it) {
        int e  = tid + 128 * it;
        int r  = e >> 10;
        int c4 = (e & 1023) << 2;
        float4 v = xg[(size_t)r * (INPUT_DIM / 4) + (e & 1023)];
        u16x4 p;
        p.x = f32_to_bf16(v.x); p.y = f32_to_bf16(v.y);
        p.z = f32_to_bf16(v.z); p.w = f32_to_bf16(v.w);
        *(u16x4*)&xs[r * XS_STRIDE + c4] = p;
    }
    for (int e = tid; e < 16 * TOPK; e += 128) {
        topv[e] = -__builtin_inff();
        topi[e] = 0;
    }
    if (tid < 16) { rowmin[tid] = -__builtin_inff(); rowpos[tid] = 0; }
    __syncthreads();

    // WMMA lane decomposition (16-bit A 16x32 / B 32x16 layouts)
    const int nsel   = lane & 15;
    const int hiHalf = lane >> 4;
    const int asel   = hiHalf << 3;        // A lanes 16-31 start at K+8
    const int ksel   = hiHalf << 3;        // B lanes 16-31 start at K+8
    const int arow   = lane & 15;
    const unsigned bwave0 = BST0 + (unsigned)wave * BWAVE_BYTES;

    for (int h0 = 0; h0 < HIDDEN_DIM; h0 += 64) {
        const int hw = h0 + wave * 16;

        // issue K-chunk 0 into buffer 0 (wave-private rows -> no barrier needed)
        #pragma unroll
        for (int i = 0; i < 16; ++i) {
            unsigned voff = ((((unsigned)(hw + i)) << 12)) * 2u + (unsigned)lane * 16u;
            async_load_b128(bwave0 + i * BROW_BYTES + (unsigned)lane * 16u, WencB, voff);
        }

        v8f acc = {0.f, 0.f, 0.f, 0.f, 0.f, 0.f, 0.f, 0.f};
        #pragma unroll 1
        for (int kc = 0; kc < INPUT_DIM / 256; ++kc) {
            const unsigned curbase = bwave0 + (unsigned)(kc & 1) * BBUF_BYTES;
            if (kc + 1 < INPUT_DIM / 256) {
                const unsigned nxtbase = bwave0 + (unsigned)((kc + 1) & 1) * BBUF_BYTES;
                const unsigned k0 = (unsigned)(kc + 1) << 8;
                #pragma unroll
                for (int i = 0; i < 16; ++i) {
                    unsigned voff = ((((unsigned)(hw + i)) << 12) + k0) * 2u
                                  + (unsigned)lane * 16u;
                    async_load_b128(nxtbase + i * BROW_BYTES + (unsigned)lane * 16u,
                                    WencB, voff);
                }
                wait_asynccnt<16>();   // current chunk landed; 16 newest in flight
            } else {
                wait_asynccnt<0>();
            }

            const unsigned brow = curbase + (unsigned)nsel * BROW_BYTES
                                + (unsigned)ksel * 2u;
            #pragma unroll
            for (int kk = 0; kk < 256; kk += 32) {
                const int kg = (kc << 8) + kk;
                union { v16bf v; u16x8 h8[2]; } A;
                A.h8[0] = *(const u16x8*)&xs[arow * XS_STRIDE + kg + asel];
                A.h8[1] = *(const u16x8*)&xs[arow * XS_STRIDE + kg + 16 + asel];
                union { v16bf v; u16x8 h8[2]; } B;
                B.h8[0] = *(const u16x8*)(smem_raw + brow + kk * 2);
                B.h8[1] = *(const u16x8*)(smem_raw + brow + kk * 2 + 32);
                acc = __builtin_amdgcn_wmma_f32_16x16x32_bf16(
                    false, A.v, false, B.v, (short)0, acc, false, false);
            }
        }

        // Scatter C tile (+bias): VGPR v holds (M = v + 8*hiHalf, N = nsel)
        float bias = b_enc[hw + nsel];
        #pragma unroll
        for (int v = 0; v < 8; ++v) {
            tile[(v + 8 * hiHalf) * TILE_STRIDE + wave * 16 + nsel] = acc[v] + bias;
        }
        __syncthreads();

        // Streaming top-K: one thread per batch row scans 64 new candidates
        if (tid < 16) {
            float  mn = rowmin[tid];
            int    mp = rowpos[tid];
            float* tv = topv + tid * TOPK;
            int*   ti = topi + tid * TOPK;
            for (int j = 0; j < 64; ++j) {
                float cand = tile[tid * TILE_STRIDE + j];
                if (cand > mn) {
                    tv[mp] = cand; ti[mp] = h0 + j;
                    mn = tv[0]; mp = 0;
                    for (int s = 1; s < TOPK; ++s)
                        if (tv[s] < mn) { mn = tv[s]; mp = s; }
                }
            }
            rowmin[tid] = mn; rowpos[tid] = mp;
        }
        __syncthreads();
    }

    for (int e = tid; e < 16 * TOPK; e += 128) {
        int r = e >> 5, s = e & 31;
        top_vals[(size_t)(row0 + r) * TOPK + s] = topv[r * TOPK + s];
        top_idx [(size_t)(row0 + r) * TOPK + s] = topi[r * TOPK + s];
    }
}

// ---------------------------------------------------------------------------
// Kernel 2: transpose + downconvert W_dec [4096][16384] f32 -> bf16 [16384][4096]
// ---------------------------------------------------------------------------
__global__ void wdec_transpose_bf16(const float* __restrict__ W_dec,
                                    unsigned short* __restrict__ WdT) {
    __shared__ float lds[32][33];
    const int h0  = blockIdx.x * 32;
    const int i0  = blockIdx.y * 32;
    const int tid = threadIdx.x;           // 256
    {
        int lh = tid & 31;
        int lb = tid >> 5;
        #pragma unroll
        for (int r = 0; r < 32; r += 8) {
            int li = lb + r;
            lds[li][lh] = W_dec[(size_t)(i0 + li) * HIDDEN_DIM + h0 + lh];
        }
    }
    __syncthreads();
    {
        int li = tid & 31;
        int lb = tid >> 5;
        #pragma unroll
        for (int r = 0; r < 32; r += 8) {
            int lh = lb + r;
            WdT[(size_t)(h0 + lh) * INPUT_DIM + i0 + li] = f32_to_bf16(lds[li][lh]);
        }
    }
}

// ---------------------------------------------------------------------------
// Kernel 3: sparse decode. One block per batch row; 32 gathered bf16 rows of
// W_decT, fully coalesced, f32 accumulation + b_dec.
// ---------------------------------------------------------------------------
__global__ void sae_decode(const float* __restrict__ top_vals,
                           const int* __restrict__ top_idx,
                           const unsigned short* __restrict__ WdT,
                           const float* __restrict__ b_dec,
                           float* __restrict__ out) {
    __shared__ float sv[TOPK];
    __shared__ int   si[TOPK];
    const int b   = blockIdx.x;
    const int tid = threadIdx.x;           // 256
    if (tid < TOPK) {
        sv[tid] = top_vals[(size_t)b * TOPK + tid];
        si[tid] = top_idx [(size_t)b * TOPK + tid];
    }
    __syncthreads();

    float acc[16];
    #pragma unroll
    for (int j = 0; j < 8; ++j) {
        int i = 2 * tid + 512 * j;
        acc[2 * j]     = b_dec[i];
        acc[2 * j + 1] = b_dec[i + 1];
    }
    for (int s = 0; s < TOPK; ++s) {
        const float           v    = sv[s];
        const unsigned short* rowp = WdT + (size_t)si[s] * INPUT_DIM;
        #pragma unroll
        for (int j = 0; j < 8; ++j) {
            unsigned int pk = *(const unsigned int*)&rowp[2 * tid + 512 * j];
            acc[2 * j]     += v * bf16_to_f32((unsigned short)(pk & 0xFFFFu));
            acc[2 * j + 1] += v * bf16_to_f32((unsigned short)(pk >> 16));
        }
    }
    #pragma unroll
    for (int j = 0; j < 8; ++j) {
        *(float2*)&out[(size_t)b * INPUT_DIM + 2 * tid + 512 * j] =
            make_float2(acc[2 * j], acc[2 * j + 1]);
    }
}

// ---------------------------------------------------------------------------
extern "C" void kernel_launch(void* const* d_in, const int* in_sizes, int n_in,
                              void* d_out, int out_size, void* d_ws, size_t ws_size,
                              hipStream_t stream) {
    const float* x     = (const float*)d_in[0];
    const float* W_enc = (const float*)d_in[1];
    const float* b_enc = (const float*)d_in[2];
    const float* W_dec = (const float*)d_in[3];
    const float* b_dec = (const float*)d_in[4];
    float*       out   = (float*)d_out;

    // Workspace: top_vals 1MB | top_idx 1MB | W_decT bf16 128MB | W_enc bf16 128MB
    float*          top_vals = (float*)d_ws;
    int*            top_idx  = (int*)((char*)d_ws + (size_t)BATCH * TOPK * 4);
    unsigned short* WdT      = (unsigned short*)((char*)d_ws + (size_t)BATCH * TOPK * 8);
    unsigned short* WencB    = (unsigned short*)((char*)d_ws + (size_t)BATCH * TOPK * 8
                                               + (size_t)HIDDEN_DIM * INPUT_DIM * 2);

    wenc_to_bf16<<<(HIDDEN_DIM * (size_t)INPUT_DIM / 4) / 256, 256, 0, stream>>>(
        W_enc, WencB);

    const size_t enc_lds = (size_t)BST0                    // xs bf16 tile
                         + 2 * BBUF_BYTES                  // double-buffered B stage
                         + (size_t)16 * TILE_STRIDE * 4    // output tile
                         + (size_t)16 * TOPK * 4           // topv
                         + (size_t)16 * TOPK * 4           // topi
                         + 16 * 4 + 16 * 4;                // rowmin/rowpos
    sae_encode_topk<<<BATCH / 16, 128, enc_lds, stream>>>(x, WencB, b_enc,
                                                          top_vals, top_idx);

    dim3 tg(HIDDEN_DIM / 32, INPUT_DIM / 32);
    wdec_transpose_bf16<<<tg, 256, 0, stream>>>(W_dec, WdT);

    sae_decode<<<BATCH, 256, 0, stream>>>(top_vals, top_idx, WdT, b_dec, out);
}
